// Seq2Seq_27522150433371
// MI455X (gfx1250) — matrix operand order repaired and checked by reading, not compiled, not run on hardware
//
#include <hip/hip_runtime.h>

#define H_ 768
#define G_ 3072   // 4*H
#define V_ 32000
#define B_ 128
#define T_ 64
#define S_ 32

#define KSTEPS (H_ / 32)   // 24
#define LDS_PITCH 40       // elements per 32-elem row in LDS (80B, 16B aligned, conflict-free)

typedef __attribute__((ext_vector_type(16))) __bf16       v16bf;
typedef __attribute__((ext_vector_type(8)))  float        v8f;
typedef __attribute__((ext_vector_type(4)))  unsigned int v4u;

union Frag16 { v16bf bf; v4u q[2]; };
union AccF   { v8f v; float f[8]; };

__device__ __forceinline__ unsigned short f2bf(float f) {
    unsigned int u = __float_as_uint(f);
    u += 0x7fffu + ((u >> 16) & 1u);          // round-to-nearest-even
    return (unsigned short)(u >> 16);
}

__device__ __forceinline__ v4u ld128(const unsigned short* p) {
    return *reinterpret_cast<const v4u*>(p);
}

// Async copy: 16B per lane, global -> LDS, tracked by ASYNCcnt (CDNA5 ISA).
__device__ __forceinline__ void async_b128(unsigned int lds_off, const void* gptr) {
    asm volatile("global_load_async_to_lds_b128 %0, %1, off"
                 :: "v"(lds_off), "v"(gptr) : "memory");
}

// ---------------------------------------------------------------------------
// Fragment loaders (ISA §7.12.2 layouts)
// ---------------------------------------------------------------------------
// A fragment 16x32 bf16 from row-major (stride `ld` elems):
// lane<16: row=lane, K=0..7 & 16..23 ; lane>=16: row=lane-16, K=8..15 & 24..31.
template<bool SWAP>
__device__ __forceinline__ Frag16 load_a_g(const unsigned short* __restrict__ A,
                                           int m_base, int k_base, int lane) {
    Frag16 f;
    int half = lane >> 4, mm = lane & 15;
    int row = m_base + mm;
    if (SWAP) row = (row & (T_ - 1)) * B_ + (row >> 6);   // out row b*T+t -> ys row t*B+b
    const unsigned short* p = A + (size_t)row * H_ + k_base + half * 8;
    f.q[0] = ld128(p);
    f.q[1] = ld128(p + 16);
    return f;
}

// B fragment 32x16: B[k][n] = W[n_base+n][k_base+k], W row-major stride H_.
__device__ __forceinline__ Frag16 load_b_g(const unsigned short* __restrict__ W,
                                           int n_base, int k_base, int lane) {
    Frag16 f;
    int half = lane >> 4, nn = lane & 15;
    const unsigned short* p = W + (size_t)(n_base + nn) * H_ + k_base + half * 8;
    f.q[0] = ld128(p);
    f.q[1] = ld128(p + 16);
    return f;
}

// Same fragments but from an LDS tile (rows of 32 elems, pitch LDS_PITCH).
__device__ __forceinline__ Frag16 load_a_s(const unsigned short* At, int row0, int lane) {
    Frag16 f;
    int half = lane >> 4, mm = lane & 15;
    const unsigned short* p = At + (row0 + mm) * LDS_PITCH + half * 8;
    f.q[0] = ld128(p);
    f.q[1] = ld128(p + 16);
    return f;
}
__device__ __forceinline__ Frag16 load_b_s(const unsigned short* Bt, int row0, int lane) {
    Frag16 f;
    int half = lane >> 4, nn = lane & 15;
    const unsigned short* p = Bt + (row0 + nn) * LDS_PITCH + half * 8;
    f.q[0] = ld128(p);
    f.q[1] = ld128(p + 16);
    return f;
}

__device__ __forceinline__ v8f wmma_bf16(Frag16 a, Frag16 b, v8f c) {
    return __builtin_amdgcn_wmma_f32_16x16x32_bf16(false, a.bf, false, b.bf,
                                                   (short)0, c, false, false);
}

__device__ __forceinline__ float sigmoidf_(float x) {
    return 1.0f / (1.0f + __expf(-x));
}

// ---------------------------------------------------------------------------
// Async-LDS double-buffered bf16 WMMA GEMM:
//   C[m][n] = sum_k A[m][k]*W[n][k] + bias[n],  K = 768
// Workgroup 256 thr = 8 waves (2M x 4N), tile 128M x 128N; wave tile 64x32.
// Per K-step: 4 global_load_async_to_lds_b128 per thread (A,B tiles),
// s_wait_asynccnt + barrier, then 8 WMMAs per wave from LDS.
// ---------------------------------------------------------------------------
template<bool SWAP, bool NT>
__global__ __launch_bounds__(256)
void wmma_gemm(const unsigned short* __restrict__ A,
               const unsigned short* __restrict__ W,
               const float* __restrict__ bias,
               float* __restrict__ C, int N) {
    __shared__ unsigned short smA[2][128 * LDS_PITCH];
    __shared__ unsigned short smB[2][128 * LDS_PITCH];

    const int tid = threadIdx.x;
    const int lane = tid & 31, w = tid >> 5;
    const int wm = w >> 2, wn = w & 3;              // 2 x 4 waves
    const int m0 = blockIdx.y * 128;
    const int n0 = blockIdx.x * 128;

    // staging assignment: thread -> (row, two 16B chunks)
    const int srow = tid >> 1;
    const int c0 = (tid & 1) * 32;                  // byte offset of first chunk in 64B row
    int grA = m0 + srow;
    if (SWAP) grA = (grA & (T_ - 1)) * B_ + (grA >> 6);
    const char* gA = (const char*)(A + (size_t)grA * H_) + c0;
    const char* gB = (const char*)(W + (size_t)(n0 + srow) * H_) + c0;
    const unsigned int lA0 = (unsigned int)(unsigned long long)&smA[0][0] + srow * (LDS_PITCH * 2) + c0;
    const unsigned int lB0 = (unsigned int)(unsigned long long)&smB[0][0] + srow * (LDS_PITCH * 2) + c0;
    const unsigned int bufStride = 128 * LDS_PITCH * 2;   // bytes between buffers

    AccF acc[4][2];
#pragma unroll
    for (int i = 0; i < 4; i++)
#pragma unroll
        for (int j = 0; j < 2; j++)
#pragma unroll
            for (int r = 0; r < 8; r++) acc[i][j].f[r] = 0.0f;

    // prologue: stage k-step 0 into buffer 0
    {
        async_b128(lA0,      gA);
        async_b128(lA0 + 16, gA + 16);
        async_b128(lB0,      gB);
        async_b128(lB0 + 16, gB + 16);
    }

    for (int kb = 0; kb < KSTEPS; ++kb) {
        if (kb + 1 < KSTEPS) {
            const int koff = (kb + 1) * 64;         // byte offset along K
            const unsigned int lbuf = ((kb + 1) & 1) * bufStride;
            async_b128(lA0 + lbuf,      gA + koff);
            async_b128(lA0 + lbuf + 16, gA + koff + 16);
            async_b128(lB0 + lbuf,      gB + koff);
            async_b128(lB0 + lbuf + 16, gB + koff + 16);
            asm volatile("s_wait_asynccnt 0x4" ::: "memory");
        } else {
            asm volatile("s_wait_asynccnt 0x0" ::: "memory");
        }
        __syncthreads();                            // tile kb ready in all waves

        const unsigned short* At = &smA[kb & 1][0];
        const unsigned short* Bt = &smB[kb & 1][0];
        Frag16 af[4], bf[2];
#pragma unroll
        for (int i = 0; i < 4; i++) af[i] = load_a_s(At, wm * 64 + i * 16, lane);
#pragma unroll
        for (int j = 0; j < 2; j++) bf[j] = load_b_s(Bt, wn * 32 + j * 16, lane);
#pragma unroll
        for (int i = 0; i < 4; i++)
#pragma unroll
            for (int j = 0; j < 2; j++)
                acc[i][j].v = wmma_bf16(af[i], bf[j], acc[i][j].v);

        __syncthreads();                            // done reading -> buffer reusable
    }

    const int nl = lane & 15, mh = (lane >> 4) * 8;
#pragma unroll
    for (int i = 0; i < 4; i++)
#pragma unroll
        for (int j = 0; j < 2; j++) {
            int n = n0 + wn * 32 + j * 16 + nl;
            float bv = bias[n];
#pragma unroll
            for (int r = 0; r < 8; r++) {
                int m = m0 + wm * 64 + i * 16 + mh + r;
                float val = acc[i][j].f[r] + bv;
                if (NT) __builtin_nontemporal_store(val, &C[(size_t)m * N + n]);
                else    C[(size_t)m * N + n] = val;
            }
        }
}

// ---------------------------------------------------------------------------
// One LSTM time step for one layer (fused WMMA GEMM + cell); latency-bound,
// keeps the direct global-load form.
// Grid: 48 blocks (16-col slices of H), 256 thr = 8 waves (16 batch rows each).
// LAYER==0: acc <- gates_x0[t] (x-part + both biases), + h0_prev@Whh0^T
// LAYER==1: acc <- (b_ih1+b_hh1), + h0_new@Wih1^T + h1_prev@Whh1^T
// ---------------------------------------------------------------------------
template<int LAYER>
__global__ __launch_bounds__(256)
void lstm_step(const unsigned short* __restrict__ Wa,   // L0: Whh0  | L1: Wih1
               const unsigned short* __restrict__ Wb,   // L0: unused| L1: Whh1
               const float* __restrict__ gx0,           // L0: gates_x0 + t*B*G
               const float* __restrict__ bsum,          // L1: b_ih1+b_hh1
               const unsigned short* __restrict__ a_in, // L0: h0_prev | L1: h0_new
               const unsigned short* __restrict__ h_prev, // L1: h1_prev
               float* __restrict__ c,                   // [B][H], in-place
               unsigned short* __restrict__ h_out) {    // bf16 [B][H]
    int lane = threadIdx.x & 31, w = threadIdx.x >> 5;
    int m0 = w * 16;
    int j0 = blockIdx.x * 16;
    int nl = lane & 15, mh = (lane >> 4) * 8;

    AccF acc[4];
    if (LAYER == 0) {
#pragma unroll
        for (int g = 0; g < 4; g++)
#pragma unroll
            for (int r = 0; r < 8; r++) {
                int b = m0 + mh + r;
                acc[g].f[r] = gx0[(size_t)b * G_ + g * H_ + j0 + nl];
            }
    } else {
#pragma unroll
        for (int g = 0; g < 4; g++) {
            float bv = bsum[g * H_ + j0 + nl];
#pragma unroll
            for (int r = 0; r < 8; r++) acc[g].f[r] = bv;
        }
    }

    for (int k = 0; k < H_; k += 32) {
        Frag16 a = load_a_g<false>(a_in, m0, k, lane);
#pragma unroll
        for (int g = 0; g < 4; g++) {
            Frag16 b = load_b_g(Wa, g * H_ + j0, k, lane);
            acc[g].v = wmma_bf16(a, b, acc[g].v);
        }
    }
    if (LAYER == 1) {
        for (int k = 0; k < H_; k += 32) {
            Frag16 a = load_a_g<false>(h_prev, m0, k, lane);
#pragma unroll
            for (int g = 0; g < 4; g++) {
                Frag16 b = load_b_g(Wb, g * H_ + j0, k, lane);
                acc[g].v = wmma_bf16(a, b, acc[g].v);
            }
        }
    }

#pragma unroll
    for (int r = 0; r < 8; r++) {
        int b = m0 + mh + r;
        int j = j0 + nl;
        float ig = sigmoidf_(acc[0].f[r]);
        float fg = sigmoidf_(acc[1].f[r]);
        float gg = tanhf(acc[2].f[r]);
        float og = sigmoidf_(acc[3].f[r]);
        float cn = fg * c[(size_t)b * H_ + j] + ig * gg;
        c[(size_t)b * H_ + j] = cn;
        h_out[(size_t)b * H_ + j] = f2bf(og * tanhf(cn));
    }
}

// ---------------------------------------------------------------------------
// Prep kernels
// ---------------------------------------------------------------------------
__global__ void cvt_bf16(const float* __restrict__ in, unsigned short* __restrict__ out, int n) {
    int i = blockIdx.x * blockDim.x + threadIdx.x;
    if (i < n) out[i] = f2bf(in[i]);
}

__global__ void embed_xs(const int* __restrict__ trg, const float* __restrict__ emb,
                         unsigned short* __restrict__ xs) {
    int idx = blockIdx.x * blockDim.x + threadIdx.x;  // (t*B+b)*H + h
    if (idx >= T_ * B_ * H_) return;
    int h = idx % H_;
    int tb = idx / H_;
    int b = tb % B_, t = tb / B_;
    int tok = trg[b * T_ + t];
    xs[idx] = f2bf(emb[(size_t)tok * H_ + h]);
}

__global__ void init_h(const int* __restrict__ src, const float* __restrict__ bert,
                       unsigned short* __restrict__ h0buf0, unsigned short* __restrict__ ysm1,
                       float* __restrict__ c01) {
    int idx = blockIdx.x * blockDim.x + threadIdx.x;  // B*H
    if (idx >= B_ * H_) return;
    int b = idx / H_, h = idx % H_;
    unsigned short bv = f2bf(bert[(size_t)src[b * S_] * H_ + h]);
    h0buf0[idx] = bv;
    ysm1[idx] = bv;
    c01[idx] = 0.0f;            // c layer 0
    c01[idx + B_ * H_] = 0.0f;  // c layer 1
}

__global__ void bias_sum(const float* __restrict__ b_ih, const float* __restrict__ b_hh,
                         float* __restrict__ bsum) {
    int i = blockIdx.x * blockDim.x + threadIdx.x;  // 2*G
    if (i < 2 * G_) bsum[i] = b_ih[i] + b_hh[i];
}

// ---------------------------------------------------------------------------
// Workspace layout (all offsets 256B aligned)
// ---------------------------------------------------------------------------
static const size_t OFF_WOUT = 0;                                  // V*H bf16
static const size_t OFF_WIH  = OFF_WOUT + (size_t)V_ * H_ * 2;     // 2*G*H bf16
static const size_t OFF_WHH  = OFF_WIH  + (size_t)2 * G_ * H_ * 2;
static const size_t OFF_XS   = OFF_WHH  + (size_t)2 * G_ * H_ * 2; // T*B*H bf16
static const size_t OFF_GX0  = OFF_XS   + (size_t)T_ * B_ * H_ * 2;// T*B*G f32
static const size_t OFF_BSUM = OFF_GX0  + (size_t)T_ * B_ * G_ * 4;// 2*G f32
static const size_t OFF_H0   = OFF_BSUM + (size_t)2 * G_ * 4;      // 2*B*H bf16 ping-pong
static const size_t OFF_YSM1 = OFF_H0   + (size_t)2 * B_ * H_ * 2; // B*H bf16
static const size_t OFF_YS   = OFF_YSM1 + (size_t)B_ * H_ * 2;     // T*B*H bf16
static const size_t OFF_C    = OFF_YS   + (size_t)T_ * B_ * H_ * 2;// 2*B*H f32

extern "C" void kernel_launch(void* const* d_in, const int* in_sizes, int n_in,
                              void* d_out, int out_size, void* d_ws, size_t ws_size,
                              hipStream_t stream) {
    const int*   src   = (const int*)  d_in[0];
    const int*   trg   = (const int*)  d_in[1];
    const float* bert  = (const float*)d_in[2];
    const float* emb   = (const float*)d_in[3];
    const float* W_ih  = (const float*)d_in[4];
    const float* W_hh  = (const float*)d_in[5];
    const float* b_ih  = (const float*)d_in[6];
    const float* b_hh  = (const float*)d_in[7];
    const float* W_out = (const float*)d_in[8];
    const float* b_out = (const float*)d_in[9];
    float* out = (float*)d_out;

    char* ws = (char*)d_ws;
    unsigned short* Wout_bf = (unsigned short*)(ws + OFF_WOUT);
    unsigned short* Wih_bf  = (unsigned short*)(ws + OFF_WIH);
    unsigned short* Whh_bf  = (unsigned short*)(ws + OFF_WHH);
    unsigned short* xs_bf   = (unsigned short*)(ws + OFF_XS);
    float*          gx0     = (float*)         (ws + OFF_GX0);
    float*          bsum    = (float*)         (ws + OFF_BSUM);
    unsigned short* h0buf   = (unsigned short*)(ws + OFF_H0);
    unsigned short* ysm1    = (unsigned short*)(ws + OFF_YSM1);
    unsigned short* ys      = (unsigned short*)(ws + OFF_YS);
    float*          cst     = (float*)         (ws + OFF_C);

    // --- prep ---
    cvt_bf16<<<(V_ * H_) / 256, 256, 0, stream>>>(W_out, Wout_bf, V_ * H_);
    cvt_bf16<<<(2 * G_ * H_) / 256, 256, 0, stream>>>(W_ih, Wih_bf, 2 * G_ * H_);
    cvt_bf16<<<(2 * G_ * H_) / 256, 256, 0, stream>>>(W_hh, Whh_bf, 2 * G_ * H_);
    embed_xs<<<(T_ * B_ * H_) / 256, 256, 0, stream>>>(trg, emb, xs_bf);
    init_h<<<(B_ * H_) / 256, 256, 0, stream>>>(src, bert, h0buf, ysm1, cst);
    bias_sum<<<(2 * G_) / 256, 256, 0, stream>>>(b_ih, b_hh, bsum);

    // --- gates_x0 = xs @ W_ih0^T + (b_ih0 + b_hh0) : (8192 x 3072 x 768) ---
    {
        dim3 grid(G_ / 128, (T_ * B_) / 128);
        wmma_gemm<false, false><<<grid, 256, 0, stream>>>(xs_bf, Wih_bf, bsum, gx0, G_);
    }

    // --- sequential LSTM: 64 steps x 2 layers ---
    for (int t = 0; t < T_; t++) {
        unsigned short* h0_cur = h0buf + (size_t)(t & 1) * B_ * H_;
        unsigned short* h0_new = h0buf + (size_t)(1 - (t & 1)) * B_ * H_;
        lstm_step<0><<<H_ / 16, 256, 0, stream>>>(
            Whh_bf, (const unsigned short*)nullptr,
            gx0 + (size_t)t * B_ * G_, bsum,
            h0_cur, (const unsigned short*)nullptr,
            cst, h0_new);
        const unsigned short* h1_prev = (t == 0) ? ysm1 : (ys + (size_t)(t - 1) * B_ * H_);
        lstm_step<1><<<H_ / 16, 256, 0, stream>>>(
            Wih_bf + (size_t)G_ * H_, Whh_bf + (size_t)G_ * H_,
            (const float*)nullptr, bsum + G_,
            h0_new, h1_prev,
            cst + (size_t)B_ * H_, ys + (size_t)t * B_ * H_);
    }

    // --- out[b,t,v] = ys[t,b,:] . W_out[v,:] + b_out[v] : (8192 x 32000 x 768) ---
    {
        dim3 grid(V_ / 128, (T_ * B_) / 128);
        wmma_gemm<true, true><<<grid, 256, 0, stream>>>(ys, Wout_bf, b_out, out, V_);
    }
}
